// GRU_D_69226282877301
// MI455X (gfx1250) — matrix operand order repaired and checked by reading, not compiled, not run on hardware
//
#include <hip/hip_runtime.h>
#include <math.h>

typedef __bf16 bf16_t;
typedef __attribute__((ext_vector_type(16))) __bf16 v16bf;
typedef __attribute__((ext_vector_type(8)))  float  v8f;

#define CB 128   // batch
#define CS 512   // seq
#define CF 128   // features
#define CH 256   // hidden
#define NROW 65536LL  // CB*CS

__device__ __forceinline__ float sigmoidf_(float v) { return 1.f / (1.f + expf(-v)); }

// ---------------- WMMA tile helpers (bf16 16x16x32, f32 accum) ----------------
// A tile: row-major [16 x 32] at (A, lda). lane m = lane&15, half = lane>>4.
// B tile: from row-major W[N,K]; tile column n is W row (colBase+n), contiguous K.
__device__ __forceinline__ v16bf load_afrag(const bf16_t* __restrict__ ap, int half) {
  v16bf a;
#pragma unroll
  for (int e = 0; e < 8; ++e) a[e] = ap[half * 8 + e];
#pragma unroll
  for (int e = 0; e < 8; ++e) a[8 + e] = ap[16 + half * 8 + e];
  return a;
}
__device__ __forceinline__ v16bf load_bfrag(const bf16_t* __restrict__ W, int ldw,
                                            int lane) {
  v16bf b;
  const bf16_t* bp = W + (long long)(lane & 15) * ldw;
  const int hb = lane >> 4;
#pragma unroll
  for (int e = 0; e < 8; ++e) b[e] = bp[hb * 8 + e];
#pragma unroll
  for (int e = 0; e < 8; ++e) b[8 + e] = bp[16 + hb * 8 + e];
  return b;
}

__device__ __forceinline__ v8f wmma_bf16(const bf16_t* __restrict__ A, int lda,
                                         const bf16_t* __restrict__ W, int ldw,
                                         v8f acc, int lane) {
  const int half = lane >> 4, mn = lane & 15;
  v16bf a = load_afrag(A + (long long)mn * lda, half);
  v16bf b = load_bfrag(W, ldw, lane);
  return __builtin_amdgcn_wmma_f32_16x16x32_bf16(false, a, false, b, (short)0, acc, false, false);
}

// Same but A is f32 in memory (hidden state), converted on load.
__device__ __forceinline__ v8f wmma_f32A(const float* __restrict__ A, int lda,
                                         const bf16_t* __restrict__ W, int ldw,
                                         v8f acc, int lane) {
  const int half = lane >> 4, mn = lane & 15;
  v16bf a;
  const float* ap = A + (long long)mn * lda;
#pragma unroll
  for (int e = 0; e < 8; ++e) a[e] = (bf16_t)ap[half * 8 + e];
#pragma unroll
  for (int e = 0; e < 8; ++e) a[8 + e] = (bf16_t)ap[16 + half * 8 + e];
  v16bf b = load_bfrag(W, ldw, lane);
  return __builtin_amdgcn_wmma_f32_16x16x32_bf16(false, a, false, b, (short)0, acc, false, false);
}

// ---------------- small utility kernels ----------------
__global__ void k_cvt(const float* __restrict__ s, bf16_t* __restrict__ d, int n) {
  int i = blockIdx.x * blockDim.x + threadIdx.x;
  if (i < n) d[i] = (bf16_t)s[i];
}
__global__ void k_cvt_slice(const float* __restrict__ s, int ld, int col0, int ncol,
                            bf16_t* __restrict__ d, int nrow) {
  int i = blockIdx.x * blockDim.x + threadIdx.x;
  if (i < nrow * ncol) {
    int r = i / ncol, c = i % ncol;
    d[i] = (bf16_t)s[(long long)r * ld + col0 + c];
  }
}
__global__ void k_bias_add(const float* __restrict__ a, const float* __restrict__ b,
                           float* __restrict__ o, int n) {
  int i = blockIdx.x * blockDim.x + threadIdx.x;
  if (i < n) o[i] = a[i] + b[i];
}
__global__ void k_zero_f32(float* p, int n) {
  int i = blockIdx.x * blockDim.x + threadIdx.x;
  if (i < n) p[i] = 0.f;
}
__global__ void k_zero_u32(unsigned* p, int n) {
  int i = blockIdx.x * blockDim.x + threadIdx.x;
  if (i < n) p[i] = 0u;
}

// ---------------- global variance of m*x (ddof=1) ----------------
__global__ void k_var_partial(const float* __restrict__ x, const float* __restrict__ m,
                              double* __restrict__ part, int n) {
  __shared__ double s1[256], s2[256];
  int tid = threadIdx.x;
  double a = 0.0, b = 0.0;
  for (int i = blockIdx.x * blockDim.x + tid; i < n; i += gridDim.x * blockDim.x) {
    double v = (double)m[i] * (double)x[i];
    a += v; b += v * v;
  }
  s1[tid] = a; s2[tid] = b; __syncthreads();
  for (int st = 128; st > 0; st >>= 1) {
    if (tid < st) { s1[tid] += s1[tid + st]; s2[tid] += s2[tid + st]; }
    __syncthreads();
  }
  if (tid == 0) { part[blockIdx.x * 2] = s1[0]; part[blockIdx.x * 2 + 1] = s2[0]; }
}
__global__ void k_var_final(const double* __restrict__ part, float* __restrict__ dout,
                            int nblk, long long n) {
  __shared__ double s1[256], s2[256];
  int tid = threadIdx.x;
  double a = 0.0, b = 0.0;
  for (int i = tid; i < nblk; i += 256) { a += part[i * 2]; b += part[i * 2 + 1]; }
  s1[tid] = a; s2[tid] = b; __syncthreads();
  for (int st = 128; st > 0; st >>= 1) {
    if (tid < st) { s1[tid] += s1[tid + st]; s2[tid] += s2[tid + st]; }
    __syncthreads();
  }
  if (tid == 0) {
    double var = (s2[0] - s1[0] * s1[0] / (double)n) / (double)(n - 1);
    dout[0] = (float)var;
  }
}

// ---------------- per-row stats: cov[r] and ct (bf16) ----------------
__global__ void k_row_stats(const float* __restrict__ x, const float* __restrict__ m,
                            const float* __restrict__ dptr,
                            float* __restrict__ cov, bf16_t* __restrict__ ctb) {
  __shared__ float sh[6][128];
  __shared__ float covsh;
  long long r = blockIdx.x;
  int f = threadIdx.x;
  long long i = r * CF + f;
  float xv = x[i], mv = m[i];
  float mx = mv * xv, om = (1.f - mv) * xv;
  sh[0][f] = om; sh[1][f] = om * om;
  sh[2][f] = xv; sh[3][f] = xv * xv;
  sh[4][f] = mx; sh[5][f] = mx * mx;
  __syncthreads();
  for (int st = 64; st > 0; st >>= 1) {
    if (f < st) {
#pragma unroll
      for (int k = 0; k < 6; ++k) sh[k][f] += sh[k][f + st];
    }
    __syncthreads();
  }
  if (f == 0) {
    const float FF = (float)CF;
    float v1 = (sh[1][0] - sh[0][0] * sh[0][0] / FF) / (FF - 1.f);
    float v2 = (sh[3][0] - sh[2][0] * sh[2][0] / FF) / (FF - 1.f);
    float v3 = (sh[5][0] - sh[4][0] * sh[4][0] / FF) / (FF - 1.f);
    float d = dptr[0];
    float cv = v1 - v2 * (v3 + d * d);
    cov[r] = cv; covsh = cv;
  }
  __syncthreads();
  float ctv = (mv == 1.f) ? 1.f : (1.f - covsh);
  ctb[i] = (bf16_t)ctv;
}

// ---------------- bf16 WMMA GEMM with register A-reuse ----------------
// C = act(A[M,K] @ W[N,K]^T + bias). Each wave owns one 16-row strip of A,
// preloads all K/32 A-fragments once, then sweeps every 16-col tile of N
// reusing them (weights stay L2/L0 resident).
// ACT: 0=none 1=sigmoid 2=relu; OUTBF: store bf16 else f32
// REMAP: output row r=b*S+s written to row s*B+b (permute to [S,B,N]).
template <int ACT, int OUTBF, int REMAP, int K>
__global__ void k_gemm(const bf16_t* __restrict__ A, const bf16_t* __restrict__ W,
                       const float* __restrict__ bias, void* __restrict__ outp,
                       int M, int N) {
  constexpr int KC = K / 32;
  int lane = threadIdx.x & 31;
  int wave = (int)((blockIdx.x * blockDim.x + threadIdx.x) >> 5);
  int strips = M >> 4;
  if (wave >= strips) return;  // wave-uniform: EXEC stays full for WMMA
  const int half = lane >> 4, n = lane & 15;
  int rowBase = wave << 4;

  // preload A fragments for the whole strip (K/32 chunks)
  v16bf af[KC];
  const bf16_t* ap = A + ((long long)rowBase + n) * K;  // lane's A row
#pragma unroll
  for (int c = 0; c < KC; ++c) af[c] = load_afrag(ap + c * 32, half);

  int tilesN = N >> 4;
  for (int nt = 0; nt < tilesN; ++nt) {
    int colBase = nt << 4;
    float bv = bias ? bias[colBase + n] : 0.f;
    v8f acc;
#pragma unroll
    for (int v = 0; v < 8; ++v) acc[v] = bv;
#pragma unroll
    for (int c = 0; c < KC; ++c) {
      v16bf b = load_bfrag(W + (long long)colBase * K + c * 32, K, lane);
      acc = __builtin_amdgcn_wmma_f32_16x16x32_bf16(false, af[c], false, b, (short)0,
                                                    acc, false, false);
    }
#pragma unroll
    for (int v = 0; v < 8; ++v) {
      long long r = rowBase + half * 8 + v;
      int c = colBase + n;
      float val = acc[v];
      if (ACT == 1) val = 1.f / (1.f + expf(-val));
      else if (ACT == 2) val = fmaxf(val, 0.f);
      long long ro = REMAP ? ((r % CS) * (long long)CB + (r / CS)) : r;
      if (OUTBF) ((bf16_t*)outp)[ro * N + c] = (bf16_t)val;
      else       ((float*)outp)[ro * N + c] = val;
    }
  }
}

// ---------------- build [x_ut | x_st] bf16 concat ----------------
__global__ void k_build_concat(const float* __restrict__ x, const float* __restrict__ t,
                               const float* __restrict__ m, const float* __restrict__ aut,
                               const float* __restrict__ cov, bf16_t* __restrict__ cat) {
  long long i = (long long)blockIdx.x * blockDim.x + threadIdx.x;  // over NROW*CF
  long long r = i >> 7;
  int f = (int)(i & 127);
  float a = aut[i], xv = x[i], mv = m[i], tv = t[i];
  float ctv = (mv == 1.f) ? 1.f : (1.f - cov[r]);
  float dec = 1.f / logf(2.7813f + tv);
  float xut = xv * a;
  float xst = (xv * floorf(ctv - 0.5f) * dec) * (xv * floorf(a - 0.5f) * dec);
  cat[r * (2 * CF) + f] = (bf16_t)xut;
  cat[r * (2 * CF) + CF + f] = (bf16_t)xst;
}

// ---------------- persistent recurrent kernel ----------------
#define RBLOCKS 16
__device__ __forceinline__ void grid_bar(unsigned* cnt, unsigned& target) {
  __syncthreads();
  target += RBLOCKS;
  if (threadIdx.x == 0) {
    __threadfence();
    atomicAdd(cnt, 1u);
    while (atomicAdd(cnt, 0u) < target) __builtin_amdgcn_s_sleep(2);
  }
  __syncthreads();
  __threadfence();
}

__global__ void k_recurrent(const float* __restrict__ Pwall,  // [S][B][512]
                            const float* __restrict__ Pwhx,   // [S][B][256]
                            const bf16_t* __restrict__ Uall,  // [512,256]
                            const bf16_t* __restrict__ Whh,   // [256,256]
                            float* __restrict__ h,            // [128,256]
                            float* __restrict__ G,            // [128,512]
                            bf16_t* __restrict__ rh,          // [128,256]
                            unsigned* __restrict__ cnt,
                            const float* __restrict__ fc_w, const float* __restrict__ fc_b,
                            float* __restrict__ out) {
  const int tid = threadIdx.x;
  const int gtid = blockIdx.x * blockDim.x + tid;  // 0..8191
  const int gw = gtid >> 5;                        // 0..255 waves
  const int lane = tid & 31;
  const int half = lane >> 4, nn = lane & 15;
  unsigned target = 0;

  for (int s = 0; s < CS; ++s) {
    // Phase A: G = Pwall[s] + h @ Uall^T  (8x32 tiles, one per wave)
    {
      const float* Ps = Pwall + (long long)s * CB * 512;
      int rowBase = (gw >> 5) << 4;   // 0..112
      int colBase = (gw & 31) << 4;   // 0..496
      v8f acc;
#pragma unroll
      for (int v = 0; v < 8; ++v)
        acc[v] = Ps[(long long)(rowBase + half * 8 + v) * 512 + colBase + nn];
#pragma unroll
      for (int k0 = 0; k0 < CH; k0 += 32)
        acc = wmma_f32A(h + (long long)rowBase * CH + k0, CH,
                        Uall + (long long)colBase * CH + k0, CH, acc, lane);
#pragma unroll
      for (int v = 0; v < 8; ++v)
        G[(long long)(rowBase + half * 8 + v) * 512 + colBase + nn] = acc[v];
    }
    grid_bar(cnt, target);

    // Phase B: z = sigmoid(G[:, :256]) stored back into G; rh = sigmoid(G[:,256:]) * h
    for (int i = gtid; i < CB * CH; i += RBLOCKS * 512) {
      int b = i >> 8, j = i & 255;
      float zv = sigmoidf_(G[(long long)b * 512 + j]);
      float rv = sigmoidf_(G[(long long)b * 512 + CH + j]);
      G[(long long)b * 512 + j] = zv;
      rh[(long long)b * CH + j] = (bf16_t)(rv * h[(long long)b * CH + j]);
    }
    grid_bar(cnt, target);

    // Phase C: T = Pwhx[s] + rh @ Whh^T ; h = (1-z)*h + z*tanh(T)  (8x16 tiles)
    if (gw < 128) {
      const float* Ps = Pwhx + (long long)s * CB * CH;
      int rowBase = (gw >> 4) << 4;
      int colBase = (gw & 15) << 4;
      v8f acc;
#pragma unroll
      for (int v = 0; v < 8; ++v)
        acc[v] = Ps[(long long)(rowBase + half * 8 + v) * CH + colBase + nn];
#pragma unroll
      for (int k0 = 0; k0 < CH; k0 += 32)
        acc = wmma_bf16(rh + (long long)rowBase * CH + k0, CH,
                        Whh + (long long)colBase * CH + k0, CH, acc, lane);
#pragma unroll
      for (int v = 0; v < 8; ++v) {
        int b = rowBase + half * 8 + v;
        int j = colBase + nn;
        float zv = G[(long long)b * 512 + j];
        float hv = h[(long long)b * CH + j];
        h[(long long)b * CH + j] = (1.f - zv) * hv + zv * tanhf(acc[v]);
      }
    }
    grid_bar(cnt, target);
  }

  // Final FC: out[b,c] = sigmoid(h[b,:] . fc_w[c,:] + fc_b[c])
  if (blockIdx.x == 0 && tid < CB * 2) {
    int b = tid >> 1, c = tid & 1;
    float sacc = fc_b[c];
    for (int j = 0; j < CH; ++j) sacc += h[(long long)b * CH + j] * fc_w[c * CH + j];
    out[b * 2 + c] = sigmoidf_(sacc);
  }
}

// ---------------- host launcher ----------------
extern "C" void kernel_launch(void* const* d_in, const int* in_sizes, int n_in,
                              void* d_out, int out_size, void* d_ws, size_t ws_size,
                              hipStream_t stream) {
  (void)in_sizes; (void)n_in; (void)out_size; (void)ws_size;
  const float* t      = (const float*)d_in[0];
  const float* x      = (const float*)d_in[1];
  const float* m      = (const float*)d_in[2];
  const float* ct_w   = (const float*)d_in[3];
  const float* ct_b   = (const float*)d_in[4];
  const float* xa_w   = (const float*)d_in[5];
  const float* xa_b   = (const float*)d_in[6];
  const float* wall_w = (const float*)d_in[7];
  const float* wall_b = (const float*)d_in[8];
  const float* uall_w = (const float*)d_in[9];
  const float* uall_b = (const float*)d_in[10];
  const float* wh_w   = (const float*)d_in[11];
  const float* wh_b   = (const float*)d_in[12];
  const float* fc_w   = (const float*)d_in[13];
  const float* fc_b   = (const float*)d_in[14];

  char* ws = (char*)d_ws;
  size_t off = 0;
  auto alloc = [&](size_t b) { size_t r = off; off += (b + 255) & ~(size_t)255; return r; };
  size_t o_part = alloc(2048 * 2 * sizeof(double));
  size_t o_d    = alloc(256);
  size_t o_cnt  = alloc(256);
  size_t o_cov  = alloc(NROW * 4);
  size_t o_ctw  = alloc(128 * 128 * 2);
  size_t o_xaw  = alloc(128 * 256 * 2);
  size_t o_wallw= alloc(512 * 128 * 2);
  size_t o_uallw= alloc(512 * 256 * 2);
  size_t o_whx  = alloc(256 * 128 * 2);
  size_t o_whh  = alloc(256 * 256 * 2);
  size_t o_wb   = alloc(512 * 4);
  size_t o_ctb  = alloc(NROW * 128 * 2);   // reused for x_adj after GEMM1
  size_t o_aut  = alloc(NROW * 128 * 4);
  size_t o_cat  = alloc(NROW * 256 * 2);
  size_t o_pwall= alloc((size_t)CS * CB * 512 * 4);
  size_t o_pwhx = alloc((size_t)CS * CB * 256 * 4);
  size_t o_h    = alloc(128 * 256 * 4);
  size_t o_G    = alloc(128 * 512 * 4);
  size_t o_rh   = alloc(128 * 256 * 2);

  // weight conversions (bf16) + fused bias
  k_cvt<<<(16384 + 255) / 256, 256, 0, stream>>>(ct_w, (bf16_t*)(ws + o_ctw), 16384);
  k_cvt<<<(32768 + 255) / 256, 256, 0, stream>>>(xa_w, (bf16_t*)(ws + o_xaw), 32768);
  k_cvt<<<(65536 + 255) / 256, 256, 0, stream>>>(wall_w, (bf16_t*)(ws + o_wallw), 65536);
  k_cvt<<<(131072 + 255) / 256, 256, 0, stream>>>(uall_w, (bf16_t*)(ws + o_uallw), 131072);
  k_cvt_slice<<<(32768 + 255) / 256, 256, 0, stream>>>(wh_w, 384, 0, 128, (bf16_t*)(ws + o_whx), 256);
  k_cvt_slice<<<(65536 + 255) / 256, 256, 0, stream>>>(wh_w, 384, 128, 256, (bf16_t*)(ws + o_whh), 256);
  k_bias_add<<<2, 256, 0, stream>>>(wall_b, uall_b, (float*)(ws + o_wb), 512);

  // global variance d
  k_var_partial<<<2048, 256, 0, stream>>>(x, m, (double*)(ws + o_part), 8388608);
  k_var_final<<<1, 256, 0, stream>>>((double*)(ws + o_part), (float*)(ws + o_d), 2048, 8388608LL);

  // per-row cov + ct(bf16)
  k_row_stats<<<65536, 128, 0, stream>>>(x, m, (float*)(ws + o_d),
                                         (float*)(ws + o_cov), (bf16_t*)(ws + o_ctb));

  // 4096 row-strips, 8 waves/block -> 512 blocks for each big GEMM
  // GEMM1: a_ut = sigmoid(ct @ ct_w^T + ct_b) -> f32
  k_gemm<1, 0, 0, 128><<<512, 256, 0, stream>>>((bf16_t*)(ws + o_ctb), (bf16_t*)(ws + o_ctw),
                                                ct_b, (void*)(ws + o_aut), 65536, 128);
  // elementwise: build [x_ut | x_st] bf16
  k_build_concat<<<32768, 256, 0, stream>>>(x, t, m, (float*)(ws + o_aut),
                                            (float*)(ws + o_cov), (bf16_t*)(ws + o_cat));
  // GEMM2: x_adj = relu(cat @ xa_w^T + xa_b) -> bf16 (reuse ct region)
  k_gemm<2, 1, 0, 256><<<512, 256, 0, stream>>>((bf16_t*)(ws + o_cat), (bf16_t*)(ws + o_xaw),
                                                xa_b, (void*)(ws + o_ctb), 65536, 128);
  // GEMM3: P_wall = x_adj @ wall_w^T + (wall_b+uall_b) -> f32 [S,B,512]
  k_gemm<0, 0, 1, 128><<<512, 256, 0, stream>>>((bf16_t*)(ws + o_ctb), (bf16_t*)(ws + o_wallw),
                                                (float*)(ws + o_wb), (void*)(ws + o_pwall),
                                                65536, 512);
  // GEMM4: P_whx = x_adj @ wh_w[:, :128]^T + wh_b -> f32 [S,B,256]
  k_gemm<0, 0, 1, 128><<<512, 256, 0, stream>>>((bf16_t*)(ws + o_ctb), (bf16_t*)(ws + o_whx),
                                                wh_b, (void*)(ws + o_pwhx), 65536, 256);

  // init state + barrier counter, then persistent recurrence + final FC
  k_zero_f32<<<128, 256, 0, stream>>>((float*)(ws + o_h), 32768);
  k_zero_u32<<<1, 32, 0, stream>>>((unsigned*)(ws + o_cnt), 32);
  k_recurrent<<<RBLOCKS, 512, 0, stream>>>((const float*)(ws + o_pwall), (const float*)(ws + o_pwhx),
                                           (const bf16_t*)(ws + o_uallw), (const bf16_t*)(ws + o_whh),
                                           (float*)(ws + o_h), (float*)(ws + o_G), (bf16_t*)(ws + o_rh),
                                           (unsigned*)(ws + o_cnt), fc_w, fc_b, (float*)d_out);
}
